// CoKEModel_57406532878361
// MI455X (gfx1250) — compile-verified, matrix operand order checked by reading, not run on hardware
//
#include <hip/hip_runtime.h>
#include <math.h>

// ---------------- model constants ----------------
static constexpr int CB   = 32;     // batch
static constexpr int CSEQ = 200;
static constexpr int CTOK = 50;
static constexpr int CE   = 512;
static constexpr int CH   = 8;
static constexpr int CDH  = 64;
static constexpr int CNH  = 2048;
static constexpr int CL   = 6;
static constexpr int CVOC = 12525;
static constexpr int CEMB = 1024;

// ---------------- WMMA types ----------------
typedef __attribute__((ext_vector_type(16))) __bf16          bf16x16;
typedef __attribute__((ext_vector_type(8)))  float           floatx8;
typedef __attribute__((ext_vector_type(8)))  unsigned short  ushort8;

__device__ __forceinline__ unsigned short f32_to_bf16(float f) {
  union { float f; unsigned int u; } c; c.f = f;
  unsigned int u = c.u;
  u += 0x7FFFu + ((u >> 16) & 1u);          // round-to-nearest-even
  return (unsigned short)(u >> 16);
}

__device__ __forceinline__ bf16x16 load_frag(const unsigned short* rowPtr, int half) {
  // 16-bit A/B fragment: elements 0..7  <- K = 8*half .. 8*half+7
  //                      elements 8..15 <- K = 16+8*half .. 16+8*half+7
  union { ushort8 h[2]; bf16x16 v; } u;
  u.h[0] = *(const ushort8*)(rowPtr + half * 8);
  u.h[1] = *(const ushort8*)(rowPtr + 16 + half * 8);
  return u.v;
}

// CDNA5 async global->LDS DMA (ASYNCcnt tracked), see CDNA5 ISA
__device__ __forceinline__ void async_b128(unsigned ldsOff, unsigned long long gaddr) {
  asm volatile("global_load_async_to_lds_b128 %0, %1, off"
               :: "v"(ldsOff), "v"(gaddr) : "memory");
}
__device__ __forceinline__ void wait_async0() {
  asm volatile("s_wait_asynccnt 0x0" ::: "memory");
}

// ---------------- bf16 GEMM: C[M,N] = A[M,Kp] @ W[N,Kp]^T + bias (+act) (+resid) ----
// A, W are pre-converted bf16 with K zero-padded to Kp (multiple of 32), so the tile
// loader is completely unguarded: pure async-DMA double-buffered through LDS.
// block: 256 threads = 8 wave32 waves; tile 128(M) x 64(N) x 32(K);
// each wave owns a 32x32 C tile = 2x2 v_wmma_f32_16x16x32_bf16 accumulators.
static constexpr int TM = 128, TN = 64, TK = 32;
static constexpr int LDA_S = 40;   // shorts per LDS row (32 + 8 pad)

template<int ACT>   // 0 = none, 1 = relu, 2 = exact gelu
__global__ __launch_bounds__(256)
void gemm_bf16_kernel(const unsigned short* __restrict__ A, const unsigned short* __restrict__ W,
                      const float* __restrict__ bias, const float* __restrict__ resid,
                      float* __restrict__ C, int M, int N, int Kp,
                      long strideA, long strideC, long strideR)
{
  __shared__ unsigned short sA[2][TM * LDA_S];
  __shared__ unsigned short sW[2][TN * LDA_S];

  const int bz = blockIdx.z;
  A += (long)bz * strideA;            // strideA in ushort elements
  C += (long)bz * strideC;
  const float* R = resid ? resid + (long)bz * strideR : nullptr;

  const int m0   = blockIdx.y * TM;
  const int n0   = blockIdx.x * TN;
  const int tid  = threadIdx.x;
  const int lane = tid & 31;
  const int wave = tid >> 5;
  const int waveM = wave & 3;   // 4 waves along M
  const int waveN = wave >> 2;  // 2 waves along N
  const int lm   = lane & 15;
  const int lh   = lane >> 4;

  // 16B chunks: A tile = 512 chunks (2/lane), W tile = 256 chunks (1/lane)
  const int cA0 = tid,        rA0 = cA0 >> 2, oA0 = (cA0 & 3) * 8;
  const int cA1 = tid + 256,  rA1 = cA1 >> 2, oA1 = (cA1 & 3) * 8;
  const int rW  = tid >> 2,   oW  = (tid & 3) * 8;

  unsigned ldsA0[2], ldsA1[2], ldsW[2];
#pragma unroll
  for (int bb = 0; bb < 2; ++bb) {
    ldsA0[bb] = (unsigned)(size_t)&sA[bb][rA0 * LDA_S + oA0];
    ldsA1[bb] = (unsigned)(size_t)&sA[bb][rA1 * LDA_S + oA1];
    ldsW [bb] = (unsigned)(size_t)&sW[bb][rW  * LDA_S + oW ];
  }
  const unsigned long long gA0 = (unsigned long long)(A + (long)(m0 + rA0) * Kp + oA0);
  const unsigned long long gA1 = (unsigned long long)(A + (long)(m0 + rA1) * Kp + oA1);
  const unsigned long long gW  = (unsigned long long)(W + (long)(n0 + rW ) * Kp + oW );

  floatx8 acc[2][2];
#pragma unroll
  for (int i = 0; i < 2; ++i)
#pragma unroll
    for (int j = 0; j < 2; ++j)
      acc[i][j] = (floatx8){0.f,0.f,0.f,0.f,0.f,0.f,0.f,0.f};

  const int nk = Kp / TK;
  // prologue: DMA first K tile into buffer 0
  {
    async_b128(ldsA0[0], gA0);
    async_b128(ldsA1[0], gA1);
    async_b128(ldsW [0], gW );
  }
  for (int kt = 0; kt < nk; ++kt) {
    wait_async0();          // this wave's DMA for tile kt done
    __syncthreads();        // all waves' DMA done; all prior reads of other buffer done
    if (kt + 1 < nk) {      // overlap tile kt+1 DMA with tile kt WMMAs
      unsigned long long off = (unsigned long long)(kt + 1) * (TK * 2);
      int nb = (kt + 1) & 1;
      async_b128(ldsA0[nb], gA0 + off);
      async_b128(ldsA1[nb], gA1 + off);
      async_b128(ldsW [nb], gW  + off);
    }
    const unsigned short* bufA = sA[kt & 1];
    const unsigned short* bufW = sW[kt & 1];

    bf16x16 bfrag[2];
#pragma unroll
    for (int ni = 0; ni < 2; ++ni)
      bfrag[ni] = load_frag(&bufW[(waveN * 32 + ni * 16 + lm) * LDA_S], lh);
#pragma unroll
    for (int mi = 0; mi < 2; ++mi) {
      bf16x16 afrag = load_frag(&bufA[(waveM * 32 + mi * 16 + lm) * LDA_S], lh);
#pragma unroll
      for (int ni = 0; ni < 2; ++ni)
        acc[mi][ni] = __builtin_amdgcn_wmma_f32_16x16x32_bf16(
            false, afrag, false, bfrag[ni], (short)0, acc[mi][ni], false, false);
    }
  }

  // ---- epilogue: reg r -> M = r + 8*lh, N = lm (within 16x16 tile)
#pragma unroll
  for (int mi = 0; mi < 2; ++mi)
#pragma unroll
    for (int ni = 0; ni < 2; ++ni)
#pragma unroll
      for (int r = 0; r < 8; ++r) {
        int gm = m0 + waveM * 32 + mi * 16 + r + 8 * lh;
        int gn = n0 + waveN * 32 + ni * 16 + lm;
        if (gm < M && gn < N) {
          float v = acc[mi][ni][r];
          if (bias) v += bias[gn];
          if (ACT == 1) v = fmaxf(v, 0.f);
          else if (ACT == 2) v = 0.5f * v * (1.f + erff(v * 0.70710678118654752f));
          if (R) v += R[(long)gm * N + gn];
          C[(long)gm * N + gn] = v;
        }
      }
}

// ---------------- fp32 -> bf16 (K zero-padded to Kp) ----------------
__global__ __launch_bounds__(256)
void cvt_bf16_kernel(const float* __restrict__ in, unsigned short* __restrict__ out,
                     int K, int Kp, long total)
{
  long idx = (long)blockIdx.x * 256 + threadIdx.x;
  if (idx >= total) return;
  int  col = (int)(idx % Kp);
  long row = idx / Kp;
  out[idx] = (col < K) ? f32_to_bf16(in[row * (long)K + col]) : (unsigned short)0;
}

// ---------------- pack conv weights into (600,1024) ----------------
__global__ __launch_bounds__(256)
void pack_w600_kernel(const float* __restrict__ cw1, const float* __restrict__ cw2,
                      const float* __restrict__ cw3, float* __restrict__ W)
{
  long idx = (long)blockIdx.x * 256 + threadIdx.x;
  if (idx >= 600L * 1024) return;
  int k = (int)(idx % 1024);
  long r = idx / 1024;
  int c = (int)(r % 100);
  int s = (int)(r / 100);
  float v;
  switch (s) {
    case 0:  v = cw1[(long)c * 1024 + k];            break;
    case 1:  v = cw2[(long)c * 2048 + k];            break;
    case 2:  v = cw2[(long)c * 2048 + 1024 + k];     break;
    case 3:  v = cw3[(long)c * 3072 + k];            break;
    case 4:  v = cw3[(long)c * 3072 + 1024 + k];     break;
    default: v = cw3[(long)c * 3072 + 2048 + k];     break;
  }
  W[idx] = v;
}

// ---------------- TextCNN features via precomputed vocab projection ----------------
__global__ __launch_bounds__(128)
void textcnn_feat_kernel(const int* __restrict__ src, const float* __restrict__ proj,
                         const float* __restrict__ cb1, const float* __restrict__ cb2,
                         const float* __restrict__ cb3, float* __restrict__ feats)
{
  int seq = blockIdx.x;           // 0..6399
  int t   = threadIdx.x;
  __shared__ int ids[CTOK];
  if (t < CTOK) ids[t] = src[(long)seq * CTOK + t];
  __syncthreads();
  if (t >= 100) return;
  int c = t;
  float b1 = cb1[c], b2 = cb2[c], b3 = cb3[c];
  float m1 = 0.f, m2 = 0.f, m3 = 0.f;
  for (int p = 0; p < CTOK; ++p) {
    const float* r0 = proj + (long)ids[p] * 600;
    m1 = fmaxf(m1, r0[c] + b1);
    if (p < CTOK - 1) {
      const float* r1 = proj + (long)ids[p + 1] * 600;
      m2 = fmaxf(m2, r0[100 + c] + r1[200 + c] + b2);
      if (p < CTOK - 2) {
        const float* r2 = proj + (long)ids[p + 2] * 600;
        m3 = fmaxf(m3, r0[300 + c] + r1[400 + c] + r2[500 + c] + b3);
      }
    }
  }
  feats[(long)seq * 300 + c]       = fmaxf(m1, 0.f);
  feats[(long)seq * 300 + 100 + c] = fmaxf(m2, 0.f);
  feats[(long)seq * 300 + 200 + c] = fmaxf(m3, 0.f);
}

// ---------------- LayerNorm (in place). mode 0: plain; 1: shared affine; 2: per-row ----
__global__ __launch_bounds__(256)
void layernorm_kernel(float* __restrict__ x, const float* __restrict__ g,
                      const float* __restrict__ b, int N, int mode, int S, int gstride)
{
  int row = blockIdx.x;
  int t   = threadIdx.x;
  __shared__ float red[256];
  float* xr = x + (long)row * N;
  float s = 0.f;
  for (int i = t; i < N; i += 256) s += xr[i];
  red[t] = s; __syncthreads();
  for (int off = 128; off > 0; off >>= 1) { if (t < off) red[t] += red[t + off]; __syncthreads(); }
  float mu = red[0] / N; __syncthreads();
  float v = 0.f;
  for (int i = t; i < N; i += 256) { float d = xr[i] - mu; v += d * d; }
  red[t] = v; __syncthreads();
  for (int off = 128; off > 0; off >>= 1) { if (t < off) red[t] += red[t + off]; __syncthreads(); }
  float rs = rsqrtf(red[0] / N + 1e-12f);
  const float *gp = g, *bp = b;
  if (mode == 2) {
    long off = (long)(row / S) * gstride + (long)(row % S) * N;
    gp = g + off; bp = b + off;
  }
  for (int i = t; i < N; i += 256) {
    float y = (xr[i] - mu) * rs;
    if (mode) y = y * gp[i] + bp[i];
    xr[i] = y;
  }
}

// ---------------- masked attention: one block per (q, h, b) ----------------
__global__ __launch_bounds__(64)
void attention_kernel(const float* __restrict__ Q, const float* __restrict__ Km,
                      const float* __restrict__ V, float* __restrict__ ctx, int S)
{
  int q = blockIdx.x, h = blockIdx.y, b = blockIdx.z;
  int t = threadIdx.x;                    // 0..63
  __shared__ float sq[CDH];
  __shared__ float sc[256];
  __shared__ float red[64];
  const long rowbase = ((long)(b * S + q)) * CE + h * CDH;
  sq[t] = Q[rowbase + t];
  __syncthreads();

  bool rowFree   = (q == 0) || (b == 0 && h >= 6);
  bool headLocal = (h < 6);
  for (int k = t; k < S; k += 64) {
    bool masked;
    if (rowFree)        masked = false;
    else if (headLocal) masked = !(k >= q - 5 && k <= q + 4);
    else                masked = true;
    float s;
    if (masked) s = -1e9f;
    else {
      const float* krow = Km + ((long)(b * S + k)) * CE + h * CDH;
      float a = 0.f;
#pragma unroll 8
      for (int d = 0; d < CDH; ++d) a += sq[d] * krow[d];
      s = a * 0.125f;
    }
    sc[k] = s;
  }
  __syncthreads();

  float mx = -1e30f;
  for (int k = t; k < S; k += 64) mx = fmaxf(mx, sc[k]);
  red[t] = mx; __syncthreads();
  for (int off = 32; off > 0; off >>= 1) { if (t < off) red[t] = fmaxf(red[t], red[t + off]); __syncthreads(); }
  mx = red[0]; __syncthreads();

  float sum = 0.f;
  for (int k = t; k < S; k += 64) { float e = __expf(sc[k] - mx); sc[k] = e; sum += e; }
  red[t] = sum; __syncthreads();
  for (int off = 32; off > 0; off >>= 1) { if (t < off) red[t] += red[t + off]; __syncthreads(); }
  float inv = 1.f / red[0];

  float a = 0.f;
  for (int k = 0; k < S; ++k) a += sc[k] * V[((long)(b * S + k)) * CE + h * CDH + t];
  ctx[rowbase + t] = a * inv;
}

// ---------------- misc small kernels ----------------
__global__ __launch_bounds__(256)
void build_x101_kernel(const float* __restrict__ yln, float* __restrict__ x101)
{
  long idx = (long)blockIdx.x * 256 + threadIdx.x;
  if (idx >= (long)CB * 101 * CE) return;
  int e = (int)(idx % CE);
  long r = idx / CE;
  int s = (int)(r % 101);
  int b = (int)(r / 101);
  x101[idx] = (s == 0) ? 0.f : yln[((long)b * CSEQ + (s - 1)) * CE + e];
}

__global__ __launch_bounds__(256)
void extract_cls_kernel(const float* __restrict__ pred, float* __restrict__ clse)
{
  int idx = blockIdx.x * 256 + threadIdx.x;
  if (idx >= CB * 200) return;
  int b = idx / 200, m = idx % 200;
  clse[idx] = pred[((long)b * 101) * 200 + m];
}

__global__ __launch_bounds__(256)
void transpose_pred_kernel(const float* __restrict__ pred, float* __restrict__ At)
{
  long idx = (long)blockIdx.x * 256 + threadIdx.x;
  if (idx >= (long)CB * 200 * 100) return;
  int j = (int)(idx % 100);
  long r = idx / 100;
  int n = (int)(r % 200);
  int b = (int)(r / 200);
  At[idx] = pred[((long)b * 101 + 1 + j) * 200 + n];
}

__global__ __launch_bounds__(256)
void softmax_rows_kernel(const float* __restrict__ in, float* __restrict__ out, int N)
{
  int row = blockIdx.x, t = threadIdx.x;
  __shared__ float red[256];
  const float* xr = in + (long)row * N;
  float mx = -1e30f;
  for (int i = t; i < N; i += 256) mx = fmaxf(mx, xr[i]);
  red[t] = mx; __syncthreads();
  for (int off = 128; off > 0; off >>= 1) { if (t < off) red[t] = fmaxf(red[t], red[t + off]); __syncthreads(); }
  mx = red[0]; __syncthreads();
  float s = 0.f;
  for (int i = t; i < N; i += 256) s += __expf(xr[i] - mx);
  red[t] = s; __syncthreads();
  for (int off = 128; off > 0; off >>= 1) { if (t < off) red[t] += red[t + off]; __syncthreads(); }
  float inv = 1.f / red[0];
  for (int i = t; i < N; i += 256) out[(long)row * N + i] = __expf(xr[i] - mx) * inv;
}

__global__ __launch_bounds__(64)
void tokvec_kernel(const float* __restrict__ lssoft, const float* __restrict__ tokmat,
                   const float* __restrict__ clse, float* __restrict__ tokvec)
{
  int n = blockIdx.x, b = blockIdx.y, t = threadIdx.x;
  __shared__ float red[64];
  float s = 0.f;
  for (int m = t; m < 200; m += 64)
    s += (lssoft[(long)n * 200 + m] + 0.4f * tokmat[((long)b * 200 + n) * 200 + m]) * clse[b * 200 + m];
  red[t] = s; __syncthreads();
  for (int off = 32; off > 0; off >>= 1) { if (t < off) red[t] += red[t + off]; __syncthreads(); }
  if (t == 0) tokvec[b * 200 + n] = red[0];
}

// segment mean then softmax. segment i covers rows [i, (i+1)*100) (reference slicing).
__global__ __launch_bounds__(256)
void seg_soft_kernel(const float* __restrict__ src, float* __restrict__ out, int S, int N)
{
  int b = blockIdx.x, i = blockIdx.y, t = threadIdx.x;
  __shared__ float buf[512];
  __shared__ float red[256];
  int start = i, cnt = (i + 1) * 100 - i;
  const float* base = src + (long)b * S * N;
  for (int n = t; n < N; n += 256) {
    float s = 0.f;
    for (int r = 0; r < cnt; ++r) s += base[(long)(start + r) * N + n];
    buf[n] = s / cnt;
  }
  __syncthreads();
  float mx = -1e30f;
  for (int n = t; n < N; n += 256) mx = fmaxf(mx, buf[n]);
  red[t] = mx; __syncthreads();
  for (int off = 128; off > 0; off >>= 1) { if (t < off) red[t] = fmaxf(red[t], red[t + off]); __syncthreads(); }
  mx = red[0]; __syncthreads();
  float s = 0.f;
  for (int n = t; n < N; n += 256) { float e = __expf(buf[n] - mx); buf[n] = e; s += e; }
  red[t] = s; __syncthreads();
  for (int off = 128; off > 0; off >>= 1) { if (t < off) red[t] += red[t + off]; __syncthreads(); }
  float inv = 1.f / red[0];
  for (int n = t; n < N; n += 256)
    out[((long)b * 2 + i) * N + n] = buf[n] * inv;
}

__global__ __launch_bounds__(32)
void loss_kernel(const float* __restrict__ segc, const float* __restrict__ sega,
                 float* __restrict__ out_loss)
{
  int b = threadIdx.x;   // one batch element per lane (wave32)
  const float* c0 = segc + (long)b * 2 * 200;
  const float* a0 = sega + (long)b * 2 * 512;
  float kl[4], euc[4];
  for (int i = 0; i < 2; ++i)
    for (int j = 0; j < 2; ++j) {
      const float* c1 = c0 + i * 200;
      const float* c2 = c0 + j * 200;
      float s = 0.f;
      for (int n = 0; n < 200; ++n) {
        float x1 = c1[n] + 1e-6f, x2 = c2[n] + 1e-6f;
        s += x1 * __logf(x1 / x2);
      }
      kl[i * 2 + j] = s;
      const float* a1 = a0 + i * 512;
      const float* a2 = a0 + j * 512;   // (+1e-6 cancels in the difference)
      float e = 0.f;
      for (int n = 0; n < 512; ++n) { float d = a1[n] - a2[n]; e += d * d; }
      euc[i * 2 + j] = sqrtf(fmaxf(e, 1e-12f));
    }
  float mk = -1e30f, me = -1e30f;
  for (int q = 0; q < 4; ++q) { mk = fmaxf(mk, kl[q] * 10.f); me = fmaxf(me, euc[q]); }
  float ek[4], ee[4], sk = 0.f, se = 0.f;
  for (int q = 0; q < 4; ++q) {
    ek[q] = __expf(kl[q] * 10.f - mk); sk += ek[q];
    ee[q] = __expf(euc[q] - me);       se += ee[q];
  }
  float l = 0.f;
  for (int q = 0; q < 4; ++q) {
    float p = ek[q] / sk + 1e-6f;
    float r = ee[q] / se + 1e-6f;
    l += p * __logf(p / r);
  }
  __shared__ float red[32];
  red[b] = l; __syncthreads();
  if (b == 0) {
    float s = 0.f;
    for (int i = 0; i < 32; ++i) s += red[i];
    out_loss[0] = (s / 32.f) * 100000.f * 0.2f;
  }
}

// ---------------- workspace layout (float units) ----------------
static constexpr long O_XA      = 0;            // 3,276,800  (also x101, M<=6400)
static constexpr long O_XB      = 3276800;
static constexpr long O_Q       = 6553600;
static constexpr long O_K       = 9830400;
static constexpr long O_V       = 13107200;
static constexpr long O_CTX     = 16384000;
static constexpr long O_H1      = 19660800;     // 13,107,200
static constexpr long O_YLN     = 32768000;     // 3,276,800
static constexpr long O_POST101 = 36044800;     // 1,654,784
static constexpr long O_POSTALL = 37699584;     // 3,276,800
static constexpr long O_PRED    = 40976384;     // 646,400
static constexpr long O_PRED1   = 41622784;     // 1,280,000
static constexpr long O_AT      = 42902784;     // 640,000
static constexpr long O_TOKMAT  = 43542784;     // 1,280,000
static constexpr long O_LSSOFT  = 44822784;     // 40,000
static constexpr long O_CLSE    = 44862784;     // 6,400
static constexpr long O_TOKVEC  = 44869184;     // 6,400
static constexpr long O_SEGC    = 44875584;     // 12,800
static constexpr long O_SEGA    = 44888384;     // 32,768
// bf16 arenas (sized in shorts, includes slack for unguarded async tile overshoot)
static constexpr long O_BFA     = 44921152;     // 13,369,344 shorts (max 6400x2048 + 128-row slack)
static constexpr long O_BFW     = 51605824;     // 1,179,648 shorts (max 2048x512 + 64-row slack)
// phase-A only (temporally overlaps encoder scratch):
static constexpr long O_PROJ    = 0;            // 7,515,000
static constexpr long O_FEATS   = 7515000;      // 1,920,000
static constexpr long O_W600    = O_H1;         // 614,400

extern "C" void kernel_launch(void* const* d_in, const int* in_sizes, int n_in,
                              void* d_out, int out_size, void* d_ws, size_t ws_size,
                              hipStream_t stream)
{
  (void)in_sizes; (void)n_in; (void)out_size; (void)ws_size;
  const int*   src   = (const int*)  d_in[0];
  const float* lsim  = (const float*)d_in[1];
  const float* emb   = (const float*)d_in[2];
  const float* cw1   = (const float*)d_in[3];
  const float* cb1   = (const float*)d_in[4];
  const float* cw2   = (const float*)d_in[5];
  const float* cb2   = (const float*)d_in[6];
  const float* cw3   = (const float*)d_in[7];
  const float* cb3   = (const float*)d_in[8];
  const float* tcfw  = (const float*)d_in[9];
  const float* tcfb  = (const float*)d_in[10];
  const float* Wq    = (const float*)d_in[11];
  const float* bq    = (const float*)d_in[12];
  const float* Wk    = (const float*)d_in[13];
  const float* bk    = (const float*)d_in[14];
  const float* Wv    = (const float*)d_in[15];
  const float* bv    = (const float*)d_in[16];
  const float* Wo    = (const float*)d_in[17];
  const float* bo    = (const float*)d_in[18];
  const float* W1    = (const float*)d_in[19];
  const float* b1    = (const float*)d_in[20];
  const float* W2    = (const float*)d_in[21];
  const float* b2    = (const float*)d_in[22];
  const float* ln1g  = (const float*)d_in[23];
  const float* ln1b  = (const float*)d_in[24];
  const float* ln2g  = (const float*)d_in[25];
  const float* ln2b  = (const float*)d_in[26];
  const float* fc1w  = (const float*)d_in[27];
  const float* fc1b  = (const float*)d_in[28];
  const float* fc2w  = (const float*)d_in[29];
  const float* fc2b  = (const float*)d_in[30];
  const float* fc3w  = (const float*)d_in[31];
  const float* fc3b  = (const float*)d_in[32];
  const float* pscale= (const float*)d_in[33];
  const float* pbias = (const float*)d_in[34];
  float* out = (float*)d_out;
  float* ws  = (float*)d_ws;
  unsigned short* bfA = (unsigned short*)(ws + O_BFA);
  unsigned short* bfW = (unsigned short*)(ws + O_BFW);

  auto cvt = [&](const float* in, unsigned short* outp, int M, int K, int Kp) {
    long tot = (long)M * Kp;
    cvt_bf16_kernel<<<(unsigned)((tot + 255) / 256), 256, 0, stream>>>(in, outp, K, Kp, tot);
  };
  auto gemm = [&](const unsigned short* A, const unsigned short* Wm, const float* bias,
                  const float* resid, float* C, int M, int N, int Kp, int act,
                  int batch = 1, long sA = 0, long sC = 0, long sR = 0) {
    dim3 grid((N + TN - 1) / TN, (M + TM - 1) / TM, batch);
    dim3 blk(256);
    if (act == 0)      gemm_bf16_kernel<0><<<grid, blk, 0, stream>>>(A, Wm, bias, resid, C, M, N, Kp, sA, sC, sR);
    else if (act == 1) gemm_bf16_kernel<1><<<grid, blk, 0, stream>>>(A, Wm, bias, resid, C, M, N, Kp, sA, sC, sR);
    else               gemm_bf16_kernel<2><<<grid, blk, 0, stream>>>(A, Wm, bias, resid, C, M, N, Kp, sA, sC, sR);
  };

  // ---- Phase A: vocab projection  proj[VOC,600] = emb[VOC,1024] @ W600[600,1024]^T
  pack_w600_kernel<<<(600 * 1024 + 255) / 256, 256, 0, stream>>>(cw1, cw2, cw3, ws + O_W600);
  cvt(emb, bfA, CVOC, CEMB, CEMB);
  cvt(ws + O_W600, bfW, 600, CEMB, CEMB);
  gemm(bfA, bfW, nullptr, nullptr, ws + O_PROJ, CVOC, 600, CEMB, 0);

  // ---- Phase B: TextCNN features for all 6400 sequences
  textcnn_feat_kernel<<<CB * CSEQ, 128, 0, stream>>>(src, ws + O_PROJ, cb1, cb2, cb3, ws + O_FEATS);

  // ---- Phase C: FC 300->512 (Kp=320) then plain LayerNorm -> y_ln (B,200,512)
  cvt(ws + O_FEATS, bfA, CB * CSEQ, 300, 320);
  cvt(tcfw, bfW, CE, 300, 320);
  gemm(bfA, bfW, tcfb, nullptr, ws + O_YLN, CB * CSEQ, CE, 320, 0);
  layernorm_kernel<<<CB * CSEQ, 256, 0, stream>>>(ws + O_YLN, nullptr, nullptr, CE, 0, 0, 0);

  // ---- build x101 = [zeros ; y_ln[:, :100]]
  build_x101_kernel<<<(CB * 101 * CE + 255) / 256, 256, 0, stream>>>(ws + O_YLN, ws + O_XA);

  auto run_encoder = [&](float* X, int S) {
    int M = CB * S;
    float* Qb  = ws + O_Q;
    float* Kb  = ws + O_K;
    float* Vb  = ws + O_V;
    float* Cb  = ws + O_CTX;
    float* Xb  = ws + O_XB;
    float* H1b = ws + O_H1;
    for (int l = 0; l < CL; ++l) {
      const long wE = (long)l * CE * CE;
      cvt(X, bfA, M, CE, CE);
      cvt(Wq + wE, bfW, CE, CE, CE);
      gemm(bfA, bfW, bq + l * CE, nullptr, Qb, M, CE, CE, 0);
      cvt(Wk + wE, bfW, CE, CE, CE);
      gemm(bfA, bfW, bk + l * CE, nullptr, Kb, M, CE, CE, 0);
      cvt(Wv + wE, bfW, CE, CE, CE);
      gemm(bfA, bfW, bv + l * CE, nullptr, Vb, M, CE, CE, 0);
      attention_kernel<<<dim3(S, CH, CB), 64, 0, stream>>>(Qb, Kb, Vb, Cb, S);
      cvt(Cb, bfA, M, CE, CE);
      cvt(Wo + wE, bfW, CE, CE, CE);
      gemm(bfA, bfW, bo + l * CE, X, Xb, M, CE, CE, 0);
      layernorm_kernel<<<M, 256, 0, stream>>>(Xb, ln1g + l * CE, ln1b + l * CE, CE, 1, 0, 0);
      cvt(Xb, bfA, M, CE, CE);
      cvt(W1 + (long)l * CNH * CE, bfW, CNH, CE, CE);
      gemm(bfA, bfW, b1 + l * CNH, nullptr, H1b, M, CNH, CE, 1 /*relu*/);
      cvt(H1b, bfA, M, CNH, CNH);
      cvt(W2 + (long)l * CE * CNH, bfW, CE, CNH, CNH);
      gemm(bfA, bfW, b2 + l * CE, Xb, X, M, CE, CNH, 0);
      layernorm_kernel<<<M, 256, 0, stream>>>(X, ln2g + l * CE, ln2b + l * CE, CE, 1, 0, 0);
    }
  };

  // ---- encoder on S=101, then post + pred heads
  run_encoder(ws + O_XA, 101);
  cvt(ws + O_XA, bfA, CB * 101, CE, CE);
  cvt(fc1w, bfW, CE, CE, CE);
  gemm(bfA, bfW, fc1b, nullptr, ws + O_POST101, CB * 101, CE, CE, 2 /*gelu*/);
  layernorm_kernel<<<CB * 101, 256, 0, stream>>>(ws + O_POST101, pscale, pbias, CE, 2, 101, 300 * CE);
  cvt(ws + O_POST101, bfA, CB * 101, CE, CE);
  cvt(fc2w, bfW, 200, CE, CE);
  gemm(bfA, bfW, fc2b, nullptr, ws + O_PRED, CB * 101, 200, CE, 0);

  extract_cls_kernel<<<(CB * 200 + 255) / 256, 256, 0, stream>>>(ws + O_PRED, ws + O_CLSE);
  transpose_pred_kernel<<<(CB * 200 * 100 + 255) / 256, 256, 0, stream>>>(ws + O_PRED, ws + O_AT);
  cvt(ws + O_AT, bfA, CB * 200, 100, 128);           // (B*200, 100) -> Kp=128
  cvt(fc3w, bfW, 200, 100, 128);
  gemm(bfA, bfW, fc3b, nullptr, ws + O_TOKMAT, 200, 200, 128, 0,
       CB, 200L * 128 /*shorts*/, 200L * 200, 0);
  softmax_rows_kernel<<<200, 256, 0, stream>>>(lsim, ws + O_LSSOFT, 200);
  tokvec_kernel<<<dim3(200, CB), 64, 0, stream>>>(ws + O_LSSOFT, ws + O_TOKMAT, ws + O_CLSE, ws + O_TOKVEC);
  softmax_rows_kernel<<<CB, 256, 0, stream>>>(ws + O_TOKVEC, out, 200);           // output 0
  softmax_rows_kernel<<<CB, 256, 0, stream>>>(ws + O_CLSE, out + CB * 200, 200);  // output 1

  // ---- encoder on S=200 (in place on y_ln), post + loss path
  run_encoder(ws + O_YLN, 200);
  cvt(ws + O_YLN, bfA, CB * 200, CE, CE);
  cvt(fc1w, bfW, CE, CE, CE);
  gemm(bfA, bfW, fc1b, nullptr, ws + O_POSTALL, CB * 200, CE, CE, 2);
  layernorm_kernel<<<CB * 200, 256, 0, stream>>>(ws + O_POSTALL, pscale, pbias, CE, 2, 200, 300 * CE);
  cvt(ws + O_POSTALL, bfA, CB * 200, CE, CE);
  cvt(fc2w, bfW, 200, CE, CE);
  gemm(bfA, bfW, fc2b, nullptr, ws + O_PRED1, CB * 200, 200, CE, 0);

  seg_soft_kernel<<<dim3(CB, 2), 256, 0, stream>>>(ws + O_PRED1, ws + O_SEGC, 200, 200);
  seg_soft_kernel<<<dim3(CB, 2), 256, 0, stream>>>(ws + O_POSTALL, ws + O_SEGA, 200, 512);
  loss_kernel<<<1, 32, 0, stream>>>(ws + O_SEGC, ws + O_SEGA, out + 2 * CB * 200);
}
